// SOPACell_26697516712353
// MI455X (gfx1250) — matrix operand order repaired and checked by reading, not compiled, not run on hardware
//
#include <hip/hip_runtime.h>

// ---------------------------------------------------------------------------
// SOPA cell for MI455X (gfx1250, wave32, WMMA).
// Pipeline: cvt(x) -> transpose weights -> GEMM1(fused gates)
//           -> segmented affine scan (3 passes) -> GEMM2
// COEF2 == 0.0 in the reference, so the c2s @ weight_out2 GEMM is skipped.
// ---------------------------------------------------------------------------

typedef __attribute__((ext_vector_type(16))) _Float16 v16h;
typedef __attribute__((ext_vector_type(8)))  _Float16 v8h;
typedef __attribute__((ext_vector_type(4)))  _Float16 v4h;
typedef __attribute__((ext_vector_type(8)))  float    v8f;

#define SEQ_L    2048
#define N_IN     1024
#define N_OUT    1024
#define MROWS    16384      // L*B
#define NGATES   6          // gate 6 of 7 is unused by the reference
#define NG_COLS  6144       // 6*1024
#define H_ELEMS  16777216   // L*B*N_OUT
#define NP       8192       // B*N_OUT independent scan lanes
#define NSEG     16
#define SEG_T    128        // SEQ_L / NSEG

// ---------------------------------------------------------------------------
// fp32 -> fp16 convert, 4 elements/thread
// ---------------------------------------------------------------------------
__global__ __launch_bounds__(256) void cvt_f32_f16(const float* __restrict__ in,
                                                   _Float16* __restrict__ out) {
    size_t i = ((size_t)blockIdx.x * 256 + threadIdx.x) * 4;
    float4 v = *(const float4*)(in + i);
    v4h o;
    o[0] = (_Float16)v.x; o[1] = (_Float16)v.y;
    o[2] = (_Float16)v.z; o[3] = (_Float16)v.w;
    *(v4h*)(out + i) = o;
}

// ---------------------------------------------------------------------------
// weight_in [1024][7168] fp32 (col = n*7+g)  ->  WtIn [6144][1024] fp16
// (row = g*1024+n: column-major K, gate-major columns; gate 6 dropped)
// ---------------------------------------------------------------------------
__global__ __launch_bounds__(256) void prep_win(const float* __restrict__ w,
                                                _Float16* __restrict__ wt) {
    __shared__ float tile[32][33];
    int k0 = blockIdx.x * 32;
    int m0 = blockIdx.y * 32;
    int tx = threadIdx.x;
    int ty = threadIdx.y;
    for (int r = ty; r < 32; r += 8)
        tile[r][tx] = w[(size_t)(k0 + r) * 7168 + m0 + tx];
    __syncthreads();
    for (int r = ty; r < 32; r += 8) {
        int m = m0 + r;
        int g = m % 7;
        int n = m / 7;
        if (g < NGATES)
            wt[(size_t)(g * N_OUT + n) * N_IN + k0 + tx] = (_Float16)tile[tx][r];
    }
}

// ---------------------------------------------------------------------------
// weight_out1 [1024][1024] fp32 -> WtOut [1024(col)][1024(k)] fp16
// ---------------------------------------------------------------------------
__global__ __launch_bounds__(256) void prep_wout(const float* __restrict__ w,
                                                 _Float16* __restrict__ wt) {
    __shared__ float tile[32][33];
    int k0 = blockIdx.x * 32;
    int m0 = blockIdx.y * 32;
    int tx = threadIdx.x;
    int ty = threadIdx.y;
    for (int r = ty; r < 32; r += 8)
        tile[r][tx] = w[(size_t)(k0 + r) * N_OUT + m0 + tx];
    __syncthreads();
    for (int r = ty; r < 32; r += 8)
        wt[(size_t)(m0 + r) * N_OUT + k0 + tx] = (_Float16)tile[tx][r];
}

// ---------------------------------------------------------------------------
// fp16 GEMM, fp32 accumulate: C[M x N] = A[M x K] * Bt[N x K]^T
//   block = 256 threads = 8 waves arranged 2(M) x 4(N)
//   wave tile = 64 x 64 (4 x 4 fragments), block tile = 128 x 256
//   16 WMMAs per 32-wide k-step, 16 b128 loads -> 1.0 load/WMMA
// Fragment layouts per CDNA5 ISA 7.12.2 (f16 16x16x32):
//   A: lane L row = L&15; halves K = (L>>4)*8 + {0..7} and +16
//   B: lane L col = L&15; halves K = (L>>4)*8 + {0..7} and +16 (same as A)
//   C: VGPR r -> row (L>>4)*8 + r, col L&15
// ---------------------------------------------------------------------------
__global__ __launch_bounds__(256) void gemm_f16(const _Float16* __restrict__ A,
                                                const _Float16* __restrict__ Bt,
                                                int Kdim, int Ncols, int mode,
                                                const float* __restrict__ bias,
                                                _Float16* __restrict__ gatesOut,
                                                float* __restrict__ hOut) {
    const int wave  = threadIdx.x >> 5;
    const int lane  = threadIdx.x & 31;
    const int l15   = lane & 15;
    const int lhalf = lane >> 4;
    const int waveM = wave >> 2;     // 0..1
    const int waveN = wave & 3;      // 0..3
    const int m0 = blockIdx.x * 128 + waveM * 64;
    const int n0 = blockIdx.y * 256 + waveN * 64;

    v8f acc[4][4] = {};

    const _Float16* Abase = A  + (size_t)(m0 + l15) * Kdim + lhalf * 8;
    const _Float16* Bbase = Bt + (size_t)(n0 + l15) * Kdim + lhalf * 8;

    for (int k = 0; k < Kdim; k += 32) {
        v16h a[4], b[4];
#pragma unroll
        for (int i = 0; i < 4; ++i) {
            const _Float16* p = Abase + (size_t)i * 16 * Kdim + k;
            v8h c0 = *(const v8h*)(p);
            v8h c1 = *(const v8h*)(p + 16);
            a[i] = __builtin_shufflevector(c0, c1, 0, 1, 2, 3, 4, 5, 6, 7,
                                                   8, 9, 10, 11, 12, 13, 14, 15);
        }
#pragma unroll
        for (int j = 0; j < 4; ++j) {
            const _Float16* p = Bbase + (size_t)j * 16 * Kdim + k;
            v8h d0 = *(const v8h*)(p);
            v8h d1 = *(const v8h*)(p + 16);
            b[j] = __builtin_shufflevector(d0, d1, 0, 1, 2, 3, 4, 5, 6, 7,
                                                   8, 9, 10, 11, 12, 13, 14, 15);
        }
#pragma unroll
        for (int i = 0; i < 4; ++i)
#pragma unroll
            for (int j = 0; j < 4; ++j)
                acc[i][j] = __builtin_amdgcn_wmma_f32_16x16x32_f16(
                    false, a[i], false, b[j], (short)0, acc[i][j], false, false);
    }

    // Epilogue
#pragma unroll
    for (int i = 0; i < 4; ++i) {
#pragma unroll
        for (int j = 0; j < 4; ++j) {
            const int col = n0 + j * 16 + l15;
            const float bv = bias[col];
#pragma unroll
            for (int r = 0; r < 8; ++r) {
                const int row = m0 + i * 16 + lhalf * 8 + r;
                float v = acc[i][j][r] + bv;
                if (mode == 0) {
                    const int g = col >> 10;          // gate index
                    const int n = col & 1023;
                    const float act = (g < 3) ? tanhf(v)
                                              : 1.0f / (1.0f + __expf(-v));
                    gatesOut[((size_t)row * N_OUT + n) * 8 + g] = (_Float16)act;
                } else {
                    hOut[(size_t)row * Ncols + col] = tanhf(v);
                }
            }
        }
    }
}

// ---------------------------------------------------------------------------
// Scan pass A: 16 parallel segments of 128 steps, zero-init carries.
// Per step stores c1hat (fp16, into c1s buffer) and A = running prod(f1).
// c2 is affine in (c2_in, d_in):  c2_t = c2hat_t + P_t*c2_in + R_t*d_in,
// with R_t = f2*R + (1-f2)*x3*D_{t-1},  d affine via (dhat, D).
// Segment-end summaries (7 x fp32) go to segsum.
// ---------------------------------------------------------------------------
__global__ __launch_bounds__(256) void scan_passA(const _Float16* __restrict__ gates,
                                                  _Float16* __restrict__ c1hatBuf,
                                                  _Float16* __restrict__ Abuf,
                                                  float* __restrict__ segsum) {
    const int tid = blockIdx.x * 256 + threadIdx.x;   // 0 .. NSEG*NP-1
    const int s = tid / NP;
    const int p = tid % NP;
    const int lbase = s * SEG_T;
    const _Float16* gp = gates + ((size_t)lbase * NP + p) * 8;

    float c1h = 0.0f, A = 1.0f;
    float dh = 0.0f, D = 1.0f;
    float c2h = 0.0f, P = 1.0f, R = 0.0f;

    constexpr int SU = 32;
    for (int t0 = 0; t0 < SEG_T; t0 += SU) {
        v8h gbuf[SU];
#pragma unroll
        for (int u = 0; u < SU; ++u)
            gbuf[u] = *(const v8h*)(gp + (size_t)(t0 + u) * (NP * 8));
#pragma unroll
        for (int u = 0; u < SU; ++u) {
            const float x1 = (float)gbuf[u][0];
            const float x2 = (float)gbuf[u][1];
            const float x3 = (float)gbuf[u][2];
            const float f1 = (float)gbuf[u][3];
            const float f2 = (float)gbuf[u][4];
            const float sl = (float)gbuf[u][5];
            // c2 pieces use dhat/D from BEFORE the d update
            const float w = (1.0f - f2) * x3;
            c2h = f2 * c2h + w * dh;
            R   = f2 * R   + w * D;
            P   = f2 * P;
            c1h = (c1h - x1) * f1 + x1;
            A   = A * f1;
            dh  = (dh - x2) * sl + x2;
            D   = D * sl;
            const size_t idx = (size_t)(lbase + t0 + u) * NP + p;
            c1hatBuf[idx] = (_Float16)c1h;
            Abuf[idx]     = (_Float16)A;
        }
    }
    const int q = s * NP + p;
    segsum[0 * (NSEG * NP) + q] = A;
    segsum[1 * (NSEG * NP) + q] = c1h;
    segsum[2 * (NSEG * NP) + q] = D;
    segsum[3 * (NSEG * NP) + q] = dh;
    segsum[4 * (NSEG * NP) + q] = P;
    segsum[5 * (NSEG * NP) + q] = R;
    segsum[6 * (NSEG * NP) + q] = c2h;
}

// ---------------------------------------------------------------------------
// Scan pass B: chain the 16 segment summaries per lane p (trivial sequential
// loop), emit incoming c1 per segment (for the fixup) and final carries.
// ---------------------------------------------------------------------------
__global__ __launch_bounds__(256) void scan_passB(const float* __restrict__ segsum,
                                                  float* __restrict__ c1in,
                                                  float* __restrict__ out) {
    const int p = blockIdx.x * 256 + threadIdx.x;   // 0..NP-1
    float c1 = 0.0f, c2 = 0.0f, d = 0.0f;
#pragma unroll
    for (int s = 0; s < NSEG; ++s) {
        c1in[s * NP + p] = c1;
        const int q = s * NP + p;
        const float A   = segsum[0 * (NSEG * NP) + q];
        const float c1h = segsum[1 * (NSEG * NP) + q];
        const float D   = segsum[2 * (NSEG * NP) + q];
        const float dh  = segsum[3 * (NSEG * NP) + q];
        const float P   = segsum[4 * (NSEG * NP) + q];
        const float R   = segsum[5 * (NSEG * NP) + q];
        const float c2h = segsum[6 * (NSEG * NP) + q];
        c2 = c2h + P * c2 + R * d;    // uses old c2, old d
        c1 = c1h + A * c1;
        d  = dh + D * d;
    }
    out[H_ELEMS + p]          = c1;   // c1f
    out[H_ELEMS + NP + p]     = c2;   // c2f
    out[H_ELEMS + 2 * NP + p] = d;    // df
}

// ---------------------------------------------------------------------------
// Scan pass C: in-place fixup  c1s[l,p] = c1hat[l,p] + A[l,p] * c1in[l/128][p]
// 8 consecutive p per thread (v8h loads/stores).
// ---------------------------------------------------------------------------
__global__ __launch_bounds__(256) void scan_passC(_Float16* __restrict__ c1s,
                                                  const _Float16* __restrict__ Abuf,
                                                  const float* __restrict__ c1in) {
    const size_t base = ((size_t)blockIdx.x * 256 + threadIdx.x) * 8;
    const int l = (int)(base / NP);
    const int p = (int)(base % NP);
    const int s = l / SEG_T;
    v8h ch = *(const v8h*)(c1s + base);
    v8h av = *(const v8h*)(Abuf + base);
    const float4 i0 = *(const float4*)(c1in + s * NP + p);
    const float4 i1 = *(const float4*)(c1in + s * NP + p + 4);
    float ini[8] = { i0.x, i0.y, i0.z, i0.w, i1.x, i1.y, i1.z, i1.w };
    v8h o;
#pragma unroll
    for (int e = 0; e < 8; ++e)
        o[e] = (_Float16)((float)ch[e] + (float)av[e] * ini[e]);
    *(v8h*)(c1s + base) = o;
}

// ---------------------------------------------------------------------------
extern "C" void kernel_launch(void* const* d_in, const int* in_sizes, int n_in,
                              void* d_out, int out_size, void* d_ws, size_t ws_size,
                              hipStream_t stream) {
    (void)in_sizes; (void)n_in; (void)out_size; (void)ws_size;
    const float* x        = (const float*)d_in[0];
    const float* win      = (const float*)d_in[1];
    const float* wout1    = (const float*)d_in[2];
    /* d_in[3] = weight_out2: unused, COEF2 == 0 */
    const float* bias_in  = (const float*)d_in[4];
    const float* bias_out = (const float*)d_in[5];
    float* out = (float*)d_out;

    // Workspace carving
    char* ws = (char*)d_ws;
    _Float16* xh     = (_Float16*)(ws);                   //  33.6 MB
    _Float16* wtin   = (_Float16*)(ws + 33554432ull);     //  12.6 MB
    _Float16* wtout  = (_Float16*)(ws + 46137344ull);     //   2.1 MB
    _Float16* gates  = (_Float16*)(ws + 48234496ull);     // 268.4 MB
    _Float16* c1s    = (_Float16*)(ws + 316669952ull);    //  33.6 MB (c1hat -> c1s)
    _Float16* Abuf   = (_Float16*)(ws + 350224384ull);    //  33.6 MB
    float*    segsum = (float*)   (ws + 383778816ull);    //   3.7 MB
    float*    c1in   = (float*)   (ws + 387448832ull);    //   0.5 MB

    // 1) x -> fp16
    cvt_f32_f16<<<H_ELEMS / (256 * 4), 256, 0, stream>>>(x, xh);

    // 2) weight transposes/permutes -> fp16, column-major K
    prep_win <<<dim3(N_IN / 32, 7168 / 32), dim3(32, 8), 0, stream>>>(win, wtin);
    prep_wout<<<dim3(N_OUT / 32, N_OUT / 32), dim3(32, 8), 0, stream>>>(wout1, wtout);

    // 3) GEMM1 (16384 x 6144 x 1024) fused with gate activations
    gemm_f16<<<dim3(MROWS / 128, NG_COLS / 256), 256, 0, stream>>>(
        xh, wtin, N_IN, NG_COLS, /*mode=*/0, bias_in, gates, nullptr);

    // 4) segmented affine scan
    scan_passA<<<(NSEG * NP) / 256, 256, 0, stream>>>(gates, c1s, Abuf, segsum);
    scan_passB<<<NP / 256, 256, 0, stream>>>(segsum, c1in, out);
    scan_passC<<<H_ELEMS / (256 * 8), 256, 0, stream>>>(c1s, Abuf, c1in);

    // 5) GEMM2 (16384 x 1024 x 1024) fused with bias + tanh -> h
    gemm_f16<<<dim3(MROWS / 128, N_OUT / 256), 256, 0, stream>>>(
        c1s, wtout, N_OUT, N_OUT, /*mode=*/1, bias_out, nullptr, out);
}
